// SuperGlue_43585328120245
// MI455X (gfx1250) — matrix operand order repaired and checked by reading, not compiled, not run on hardware
//
#include <hip/hip_runtime.h>

// SuperGlue forward for MI455X (gfx1250, wave32, WMMA).
// All matmuls: double-buffered LDS-staged block-tile GEMM ->
// v_wmma_f32_16x16x32_bf16 (bf16 operands, fp32 accumulate).
// Working set (~85 MB) is L2-resident (192 MB): HBM is irrelevant, so the
// win is overlapping L2->LDS staging with WMMA. Direct-layout staging uses
// CDNA5 async global->LDS loads (ASYNCcnt + s_wait_asynccnt); transposed
// staging scatters via ds_store_b16. Fragments are conflict-free
// ds_load_b128 from canonical Al[m][k] / Bl[n][k] tiles (stride 40).

#define BB   2
#define NN   1024
#define DD   256
#define HH   4
#define HEADD 64
#define LL   9
#define M1   1025   // N+1 (Sinkhorn augmented size)

#define ASTR 40     // LDS row stride (elements) for A tile [64][ASTR]
#define BSTR 40     // LDS row stride (elements) for B tile [32][BSTR]

typedef __attribute__((ext_vector_type(16))) __bf16 v16bf;
typedef __attribute__((ext_vector_type(8)))  float  v8f;
typedef __attribute__((ext_vector_type(8)))  __bf16 bf16x8;
typedef __attribute__((ext_vector_type(4)))  __bf16 bf16x4;

// ---------------- block reduction helpers (blockDim == 256) ----------------
__device__ __forceinline__ float blk_sum(float val, float* red) {
  int t = threadIdx.x;
  red[t] = val; __syncthreads();
  for (int s = 128; s > 0; s >>= 1) { if (t < s) red[t] += red[t + s]; __syncthreads(); }
  float r = red[0]; __syncthreads();
  return r;
}
__device__ __forceinline__ float blk_max(float val, float* red) {
  int t = threadIdx.x;
  red[t] = val; __syncthreads();
  for (int s = 128; s > 0; s >>= 1) { if (t < s) red[t] = fmaxf(red[t], red[t + s]); __syncthreads(); }
  float r = red[0]; __syncthreads();
  return r;
}

// ---- CDNA5 async global->LDS (ASYNCcnt-tracked, bypasses VGPRs) ----
__device__ __forceinline__ void async_b128_to_lds(unsigned lds_off, const void* gaddr) {
  asm volatile("global_load_async_to_lds_b128 %0, %1, off"
               :: "v"(lds_off), "v"(gaddr) : "memory");
}
__device__ __forceinline__ void async_b64_to_lds(unsigned lds_off, const void* gaddr) {
  asm volatile("global_load_async_to_lds_b64 %0, %1, off"
               :: "v"(lds_off), "v"(gaddr) : "memory");
}
__device__ __forceinline__ void wait_async0() {
  asm volatile("s_wait_asynccnt 0x0" ::: "memory");
}

// ---------------- double-buffered LDS-staged bf16 WMMA GEMM ----------------
// C[m,n] = alpha * sum_k A(m,k)*B(k,n) (+bias[m]); writes f32 (Cf) and/or
// bf16 (Cb) with the same element strides.
// A(m,k) at A[m*a_rs + k*a_cs]; B(k,n) at B[k*b_rs + n*b_cs].
// AM=0: a_cs==1 (K-contig rows -> async b128 direct);  AM=1: a_rs==1 (M-contig
// -> coalesced b128 into VGPRs + transposed ds_store_b16 scatter).
// BM=0: b_cs==1 (N-contig rows -> transposed scatter); BM=1: b_rs==1 (K-contig
// -> async b64 direct).
// Block: 256 threads = 8 waves; block tile 64(M) x 32(N); K-step 32.
// Pipeline: prefetch tile i+1 into buf^1 right after the barrier while
// computing tile i from buf. One barrier per iteration: a wave's fragment
// ds_loads complete (s_wait_dscnt) before its wmma, which precedes its next
// barrier signal, so post-barrier overwrites of the old buffer are safe.
// M % 64 == 0, N % 32 == 0, K % 32 == 0 (true for every call site here).
template<int AM, int BM>
__global__ __launch_bounds__(256) void gemm_tile(
    const __bf16* __restrict__ A, const __bf16* __restrict__ Bmat,
    float* __restrict__ Cf, __bf16* __restrict__ Cb,
    int M, int Nn, int K,
    long a_rs, long a_cs, long b_rs, long b_cs, long c_rs, long c_cs,
    int ZI, long a_bo, long a_bi, long b_bo, long b_bi, long c_bo, long c_bi,
    const float* __restrict__ bias, float alpha)
{
  __shared__ __align__(16) __bf16 Al[2][64 * ASTR];  // canonical [m][k]
  __shared__ __align__(16) __bf16 Bl[2][32 * BSTR];  // canonical [n][k]
  int t = threadIdx.x;
  int mb = M >> 6;
  int bm = blockIdx.x % mb, bn = blockIdx.x / mb;
  int z = blockIdx.z, zo = z / ZI, zi = z % ZI;
  A    += (long)zo * a_bo + (long)zi * a_bi;
  Bmat += (long)zo * b_bo + (long)zi * b_bi;
  long coff = (long)zo * c_bo + (long)zi * c_bi;
  int m0 = bm << 6, n0 = bn << 5;
  int w = t >> 5, lane = t & 31;
  int wm = w & 3, wn = w >> 2;
  int l15 = lane & 15, ksel = lane >> 4;

  auto stageA = [&](int k0, int buf) {
    if (AM == 0) {
      int am = t >> 2, akc = (t & 3) << 3;   // 8 bf16 (16B) along k, direct
      async_b128_to_lds((unsigned)(size_t)&Al[buf][am * ASTR + akc],
                        A + (long)(m0 + am) * a_rs + (k0 + akc));
    } else {
      int ak = t >> 3, amc = (t & 7) << 3;   // 8 bf16 (16B) along m, transpose
      bf16x8 va = *(const bf16x8*)(A + (long)(k0 + ak) * a_cs + (m0 + amc));
#pragma unroll
      for (int i = 0; i < 8; ++i) Al[buf][(amc + i) * ASTR + ak] = va[i];
    }
  };
  auto stageB = [&](int k0, int buf) {
    if (BM == 0) {
      int bk = t >> 3, bnc = (t & 7) << 2;   // 4 bf16 (8B) along n, transpose
      bf16x4 vb = *(const bf16x4*)(Bmat + (long)(k0 + bk) * b_rs + (n0 + bnc));
#pragma unroll
      for (int i = 0; i < 4; ++i) Bl[buf][(bnc + i) * BSTR + bk] = vb[i];
    } else {
      int bn_ = t >> 3, bkc = (t & 7) << 2;  // 4 bf16 (8B) along k, direct
      async_b64_to_lds((unsigned)(size_t)&Bl[buf][bn_ * BSTR + bkc],
                       Bmat + (long)(n0 + bn_) * b_cs + (k0 + bkc));
    }
  };

  const int nk = K >> 5;
  stageA(0, 0);
  stageB(0, 0);
  v8f acc = {};
  for (int i = 0; i < nk; ++i) {
    if (AM == 0 || BM == 1) wait_async0();  // own async stores visible pre-barrier
    __syncthreads();                        // (ds stores waited by compiler)
    int cur = i & 1;
    if (i + 1 < nk) {                       // prefetch next tile into other buffer
      stageA((i + 1) << 5, cur ^ 1);
      stageB((i + 1) << 5, cur ^ 1);
    }
    // ---- fragments: 4x conflict-free ds_load_b128 per wave ----
    int mloc = wm * 16 + l15, ka = ksel * 8;
    bf16x8 a_lo = *(const bf16x8*)&Al[cur][mloc * ASTR + ka];
    bf16x8 a_hi = *(const bf16x8*)&Al[cur][mloc * ASTR + ka + 16];
    int nloc = wn * 16 + l15, kb = ksel * 8;
    bf16x8 b_lo = *(const bf16x8*)&Bl[cur][nloc * BSTR + kb];
    bf16x8 b_hi = *(const bf16x8*)&Bl[cur][nloc * BSTR + kb + 16];
    v16bf av, bv;
#pragma unroll
    for (int i2 = 0; i2 < 8; ++i2) {
      av[i2] = a_lo[i2]; av[i2 + 8] = a_hi[i2];
      bv[i2] = b_lo[i2]; bv[i2 + 8] = b_hi[i2];
    }
    acc = __builtin_amdgcn_wmma_f32_16x16x32_bf16(false, av, false, bv,
                                                  (short)0, acc, false, false);
  }
  // ---- epilogue: optional bias, write f32 and/or bf16 ----
  int cn = n0 + wn * 16 + l15;
  float badd[8];
#pragma unroll
  for (int j = 0; j < 8; ++j) badd[j] = 0.f;
  if (bias) {
#pragma unroll
    for (int j = 0; j < 8; ++j) badd[j] = bias[m0 + wm * 16 + j + 8 * ksel];
  }
#pragma unroll
  for (int j = 0; j < 8; ++j) {
    int cm = m0 + wm * 16 + j + 8 * ksel;
    float v = acc[j] * alpha + badd[j];
    long idx = coff + (long)cm * c_rs + (long)cn * c_cs;
    if (Cf) Cf[idx] = v;
    if (Cb) Cb[idx] = (__bf16)v;
  }
}

// ---------------- keypoint encoder (MLP 3->32->64->128->256) ----------------
__device__ void kenc_layer(const float* __restrict__ w, const float* __restrict__ bias,
                           const float* __restrict__ g, const float* __restrict__ be,
                           int Cin, int Cout, const float* xin, float* xout, float* red)
{
  int t = threadIdx.x;
  float acc = 0.f;
  if (t < Cout) {
    acc = bias[t];
    for (int c = 0; c < Cin; ++c) acc += w[t * Cin + c] * xin[c];
  }
  if (g) {  // LayerNorm over channels (ddof=1, eps added to std) + ReLU
    float sum  = blk_sum(t < Cout ? acc : 0.f, red);
    float mean = sum / (float)Cout;
    float d    = (t < Cout) ? (acc - mean) : 0.f;
    float ss   = blk_sum(d * d, red);
    float stdv = sqrtf(ss / (float)(Cout - 1));
    if (t < Cout) {
      float y = g[t] * (acc - mean) / (stdv + 1e-6f) + be[t];
      xout[t] = fmaxf(y, 0.f);
    }
  } else {
    if (t < Cout) xout[t] = acc;
  }
  __syncthreads();
}

__global__ __launch_bounds__(256) void kenc_kernel(
    const float* __restrict__ kpts, const float* __restrict__ scores,
    const float* __restrict__ desc_in,
    const float* w0, const float* b0, const float* g0, const float* be0,
    const float* w1, const float* b1, const float* g1, const float* be1,
    const float* w2, const float* b2, const float* g2, const float* be2,
    const float* w3, const float* b3,
    float* __restrict__ descf, __bf16* __restrict__ descb)
{
  __shared__ float bufA[256], bufB[256], red[256];
  int p = blockIdx.x, b = p / NN, n = p % NN, t = threadIdx.x;
  if (t == 0) {
    float kx = kpts[((long)b * NN + n) * 2 + 0];
    float ky = kpts[((long)b * NN + n) * 2 + 1];
    bufA[0] = (kx - 320.f) / 448.f;   // center (640,480)/2, scale max(size)*0.7
    bufA[1] = (ky - 240.f) / 448.f;
    bufA[2] = scores[(long)b * NN + n];
  }
  __syncthreads();
  kenc_layer(w0, b0, g0, be0,   3,  32, bufA, bufB, red);
  kenc_layer(w1, b1, g1, be1,  32,  64, bufB, bufA, red);
  kenc_layer(w2, b2, g2, be2,  64, 128, bufA, bufB, red);
  kenc_layer(w3, b3, nullptr, nullptr, 128, 256, bufB, bufA, red);
  long idx = ((long)b * DD + t) * NN + n;
  float v = desc_in[idx] + bufA[t];
  descf[idx] = v;
  descb[idx] = (__bf16)v;
}

// ---------------- softmax over last dim, writes bf16 ----------------
__global__ __launch_bounds__(256) void softmax_bf16(const float* __restrict__ S,
                                                    __bf16* __restrict__ P, int ncols)
{
  __shared__ float red[256];
  long row = blockIdx.x;
  const float* s = S + row * (long)ncols;
  __bf16* p = P + row * (long)ncols;
  int t = threadIdx.x;
  float m = -3.0e38f;
  for (int j = t; j < ncols; j += 256) m = fmaxf(m, s[j]);
  m = blk_max(m, red);
  float sum = 0.f;
  for (int j = t; j < ncols; j += 256) sum += __expf(s[j] - m);
  sum = blk_sum(sum, red);
  float inv = 1.f / sum;
  for (int j = t; j < ncols; j += 256) p[j] = (__bf16)(__expf(s[j] - m) * inv);
}

// ---------------- LayerNorm (channel dim, ddof=1) + ReLU -> bf16 ----------------
__global__ __launch_bounds__(256) void ln_relu_bf16(const float* __restrict__ X,
    const float* __restrict__ g, const float* __restrict__ be,
    __bf16* __restrict__ Y, int Cc)
{
  __shared__ float red[256];
  int p = blockIdx.x, b = p / NN, n = p % NN, t = threadIdx.x;
  const float* x = X + (long)b * Cc * NN + n;
  float s1 = 0.f;
  for (int c = t; c < Cc; c += 256) s1 += x[(long)c * NN];
  float mean = blk_sum(s1, red) / (float)Cc;
  float s2 = 0.f;
  for (int c = t; c < Cc; c += 256) { float d = x[(long)c * NN] - mean; s2 += d * d; }
  float stdv = sqrtf(blk_sum(s2, red) / (float)(Cc - 1));
  float inv = 1.f / (stdv + 1e-6f);
  __bf16* y = Y + (long)b * Cc * NN + n;
  for (int c = t; c < Cc; c += 256) {
    float yv = g[c] * (x[(long)c * NN] - mean) * inv + be[c];
    y[(long)c * NN] = (__bf16)fmaxf(yv, 0.f);
  }
}

// ---------------- elementwise helpers ----------------
__global__ void add_update(float* __restrict__ descf, __bf16* __restrict__ descb,
                           const float* __restrict__ dx, long count)
{
  long i = (long)blockIdx.x * blockDim.x + threadIdx.x;
  if (i < count) { float v = descf[i] + dx[i]; descf[i] = v; descb[i] = (__bf16)v; }
}
__global__ void f2bf(__bf16* __restrict__ dst, const float* __restrict__ src, long count)
{
  long i = (long)blockIdx.x * blockDim.x + threadIdx.x;
  if (i < count) dst[i] = (__bf16)src[i];
}
__global__ void bcopy_b(__bf16* __restrict__ dst, const __bf16* __restrict__ src,
                        long per, long dstb, long srcb)
{
  long i = (long)blockIdx.x * blockDim.x + threadIdx.x;
  if (i < per) dst[(long)blockIdx.y * dstb + i] = src[(long)blockIdx.y * srcb + i];
}
__global__ void zero_f(float* __restrict__ p, long count)
{
  long i = (long)blockIdx.x * blockDim.x + threadIdx.x;
  if (i < count) p[i] = 0.f;
}

// ---------------- Sinkhorn ----------------
__global__ void build_Z(const float* __restrict__ sc, const float* __restrict__ alpha_p,
                        float* __restrict__ Z)
{
  long i = (long)blockIdx.x * blockDim.x + threadIdx.x;
  long total = (long)BB * M1 * M1;
  if (i >= total) return;
  long b = i / ((long)M1 * M1), r = i % ((long)M1 * M1);
  int ii = (int)(r / M1), jj = (int)(r % M1);
  float v = (ii < NN && jj < NN) ? sc[((long)b * NN + ii) * NN + jj] : *alpha_p;
  Z[i] = v;
}
__global__ __launch_bounds__(256) void sink_u(const float* __restrict__ Z,
    const float* __restrict__ v, float* __restrict__ u)
{
  __shared__ float red[256];
  int r = blockIdx.x, b = r / M1, i = r % M1, t = threadIdx.x;
  const float* z  = Z + ((long)b * M1 + i) * M1;
  const float* vv = v + (long)b * M1;
  float m = -3.0e38f;
  for (int j = t; j < M1; j += 256) m = fmaxf(m, z[j] + vv[j]);
  m = blk_max(m, red);
  float s = 0.f;
  for (int j = t; j < M1; j += 256) s += __expf(z[j] + vv[j] - m);
  s = blk_sum(s, red);
  if (t == 0) {
    float norm = -logf(2.f * NN);
    float lmu = (i < NN) ? norm : (logf((float)NN) + norm);
    u[(long)b * M1 + i] = lmu - (m + logf(s));
  }
}
__global__ __launch_bounds__(256) void sink_v(const float* __restrict__ Z,
    const float* __restrict__ u, float* __restrict__ v)
{
  __shared__ float red[256];
  int r = blockIdx.x, b = r / M1, j = r % M1, t = threadIdx.x;
  const float* z  = Z + (long)b * M1 * M1 + j;
  const float* uu = u + (long)b * M1;
  float m = -3.0e38f;
  for (int i = t; i < M1; i += 256) m = fmaxf(m, z[(long)i * M1] + uu[i]);
  m = blk_max(m, red);
  float s = 0.f;
  for (int i = t; i < M1; i += 256) s += __expf(z[(long)i * M1] + uu[i] - m);
  s = blk_sum(s, red);
  if (t == 0) {
    float norm = -logf(2.f * NN);
    float lnu = (j < NN) ? norm : (logf((float)NN) + norm);
    v[(long)b * M1 + j] = lnu - (m + logf(s));
  }
}
__global__ void sink_final(const float* __restrict__ Z, const float* __restrict__ u,
                           const float* __restrict__ v, float* __restrict__ out)
{
  long i = (long)blockIdx.x * blockDim.x + threadIdx.x;
  long total = (long)BB * M1 * M1;
  if (i >= total) return;
  long b = i / ((long)M1 * M1), r = i % ((long)M1 * M1);
  int ii = (int)(r / M1), jj = (int)(r % M1);
  out[i] = Z[i] + u[(long)b * M1 + ii] + v[(long)b * M1 + jj] + logf(2.f * NN);
}

// ---------------- host-side GEMM launcher ----------------
template<int AMODE, int BMODE>
static void launch_gemm(hipStream_t stream,
                        const __bf16* A, const __bf16* Bm, float* Cf, __bf16* Cb,
                        int M, int Nc, int K,
                        long a_rs, long a_cs, long b_rs, long b_cs, long c_rs, long c_cs,
                        int Z, int ZI, long a_bo, long a_bi, long b_bo, long b_bi,
                        long c_bo, long c_bi, const float* bias, float alpha)
{
  dim3 grid((unsigned)((M / 64) * (Nc / 32)), 1, (unsigned)Z);
  gemm_tile<AMODE, BMODE><<<grid, 256, 0, stream>>>(A, Bm, Cf, Cb, M, Nc, K,
      a_rs, a_cs, b_rs, b_cs, c_rs, c_cs,
      ZI, a_bo, a_bi, b_bo, b_bi, c_bo, c_bi, bias, alpha);
}

// ---------------- host orchestration ----------------
extern "C" void kernel_launch(void* const* d_in, const int* in_sizes, int n_in,
                              void* d_out, int out_size, void* d_ws, size_t ws_size,
                              hipStream_t stream)
{
  (void)in_sizes; (void)n_in; (void)out_size; (void)ws_size;
  // inputs per setup_inputs() order
  const float* kpts0   = (const float*)d_in[0];
  const float* kpts1   = (const float*)d_in[1];
  const float* scores0 = (const float*)d_in[2];
  const float* scores1 = (const float*)d_in[3];
  const float* desc0_in = (const float*)d_in[4];
  const float* desc1_in = (const float*)d_in[5];
  const float* kw0 = (const float*)d_in[6];  const float* kb0 = (const float*)d_in[7];
  const float* kg0 = (const float*)d_in[8];  const float* kbe0 = (const float*)d_in[9];
  const float* kw1 = (const float*)d_in[10]; const float* kb1 = (const float*)d_in[11];
  const float* kg1 = (const float*)d_in[12]; const float* kbe1 = (const float*)d_in[13];
  const float* kw2 = (const float*)d_in[14]; const float* kb2 = (const float*)d_in[15];
  const float* kg2 = (const float*)d_in[16]; const float* kbe2 = (const float*)d_in[17];
  const float* kw3 = (const float*)d_in[18]; const float* kb3 = (const float*)d_in[19];
  const float* gnn_proj_w  = (const float*)d_in[20];
  const float* gnn_proj_b  = (const float*)d_in[21];
  const float* gnn_merge_w = (const float*)d_in[22];
  const float* gnn_merge_b = (const float*)d_in[23];
  const float* gnn_mlp1_w  = (const float*)d_in[24];
  const float* gnn_mlp1_b  = (const float*)d_in[25];
  const float* gnn_ln_g    = (const float*)d_in[26];
  const float* gnn_ln_b    = (const float*)d_in[27];
  const float* gnn_mlp2_w  = (const float*)d_in[28];
  const float* gnn_mlp2_b  = (const float*)d_in[29];
  const float* final_w     = (const float*)d_in[30];
  const float* final_b     = (const float*)d_in[31];
  const float* bin_score   = (const float*)d_in[32];

  // workspace carve (every buffer written before read each call)
  char* base = (char*)d_ws;
  size_t off = 0;
  auto carveF = [&](long n) -> float* {
    off = (off + 255) & ~(size_t)255; float* p = (float*)(base + off);
    off += (size_t)n * sizeof(float); return p;
  };
  auto carveB = [&](long n) -> __bf16* {
    off = (off + 255) & ~(size_t)255; __bf16* p = (__bf16*)(base + off);
    off += (size_t)n * sizeof(__bf16); return p;
  };
  const long DN = (long)DD * NN;
  float*  desc0f = carveF((long)BB * DN);
  float*  desc1f = carveF((long)BB * DN);
  __bf16* desc0b = carveB((long)BB * DN);
  __bf16* desc1b = carveB((long)BB * DN);
  __bf16* projb  = carveB((long)LL * 3 * DD * DD);
  __bf16* mergeb = carveB((long)LL * DD * DD);
  __bf16* mlp1b  = carveB((long)LL * 512 * 512);
  __bf16* mlp2b  = carveB((long)LL * DD * 512);
  __bf16* finalb = carveB((long)DD * DD);
  __bf16* qb     = carveB((long)BB * DN);
  __bf16* kvb    = carveB((long)BB * 512 * NN);
  float*  attnf  = carveF((long)BB * HH * NN * NN);   // 32 MB; aliased below
  __bf16* probb  = carveB((long)BB * HH * NN * NN);
  __bf16* msgb   = carveB((long)BB * DN);
  __bf16* ybufb  = carveB((long)BB * 512 * NN);
  float*  h1f    = carveF((long)BB * 512 * NN);
  __bf16* h1b    = carveB((long)BB * 512 * NN);
  float*  dx0f   = carveF((long)BB * DN);
  float*  dx1f   = carveF((long)BB * DN);
  __bf16* md0b   = carveB((long)BB * DN);
  __bf16* md1b   = carveB((long)BB * DN);
  float*  ubuf   = carveF((long)BB * M1);
  float*  vbuf   = carveF((long)BB * M1);
  // attention buffer is dead after the GNN -> alias for Sinkhorn matrices
  float*  scmat  = attnf;                              // [B,N,N] fp32
  float*  Zf     = attnf + (long)BB * NN * NN;         // [B,1025,1025] fp32

  auto cvt = [&](__bf16* dst, const float* src, long n) {
    f2bf<<<dim3((unsigned)((n + 255) / 256)), 256, 0, stream>>>(dst, src, n);
  };

  // 1) weights -> bf16
  cvt(projb,  gnn_proj_w,  (long)LL * 3 * DD * DD);
  cvt(mergeb, gnn_merge_w, (long)LL * DD * DD);
  cvt(mlp1b,  gnn_mlp1_w,  (long)LL * 512 * 512);
  cvt(mlp2b,  gnn_mlp2_w,  (long)LL * DD * 512);
  cvt(finalb, final_w,     (long)DD * DD);

  // 2) keypoint encoder + residual add into working descriptors
  kenc_kernel<<<BB * NN, 256, 0, stream>>>(kpts0, scores0, desc0_in,
      kw0, kb0, kg0, kbe0, kw1, kb1, kg1, kbe1, kw2, kb2, kg2, kbe2, kw3, kb3,
      desc0f, desc0b);
  kenc_kernel<<<BB * NN, 256, 0, stream>>>(kpts1, scores1, desc1_in,
      kw0, kb0, kg0, kbe0, kw1, kb1, kg1, kbe1, kw2, kb2, kg2, kbe2, kw3, kb3,
      desc1f, desc1b);

  // 3) GNN cross-attention layers
  auto prop = [&](int i, const __bf16* xb, const __bf16* srcb, float* dxf) {
    const __bf16* pw = projb + (long)i * 3 * DD * DD;
    const float*  pb = gnn_proj_b + (long)i * 3 * DD;
    // q = W0*x (+b0): weights K-contig (AM=0), activations N-contig (BM=0)
    launch_gemm<0,0>(stream, pw, xb, nullptr, qb, DD, NN, DD,
                     DD, 1, NN, 1, NN, 1,
                     BB, 1, 0, 0, DN, 0, DN, 0, pb, 1.f);
    // kv = [W1;W2]*src (+[b1;b2])
    launch_gemm<0,0>(stream, pw + (long)DD * DD, srcb, nullptr, kvb, 512, NN, DD,
                     DD, 1, NN, 1, NN, 1,
                     BB, 1, 0, 0, DN, 0, (long)512 * NN, 0, pb + DD, 1.f);
    // scores[b,h,n,m] = (1/8) sum_d q[b,4d+h,n]*k[b,4d+h,m]   (z = b*H+h)
    // A = q view: a_rs=1 (M-contig -> AM=1), a_cs=4N; B = k view: b_cs=1 (BM=0)
    launch_gemm<1,0>(stream, qb, kvb, attnf, nullptr, NN, NN, HEADD,
                     1, 4L * NN, 4L * NN, 1, NN, 1,
                     BB * HH, HH,
                     DN, NN, (long)512 * NN, NN,
                     (long)HH * NN * NN, (long)NN * NN,
                     nullptr, 0.125f);
    softmax_bf16<<<BB * HH * NN, 256, 0, stream>>>(attnf, probb, NN);
    // msg[b,4d+h,n] = sum_m P[b,h,n,m]*v[b,4d+h,m]; v = kv rows 256..511
    // A = v view: a_cs=1 (AM=0); B = prob: b_rs=1 (K-contig -> BM=1), b_cs=N
    launch_gemm<0,1>(stream, kvb + (long)256 * NN, probb, nullptr, msgb,
                     HEADD, NN, NN,
                     4L * NN, 1, 1, NN, 4L * NN, 1,
                     BB * HH, HH,
                     (long)512 * NN, NN, (long)HH * NN * NN, (long)NN * NN,
                     DN, NN,
                     nullptr, 1.f);
    // merge -> writes bf16 directly into second half of concat buffer
    launch_gemm<0,0>(stream, mergeb + (long)i * DD * DD, msgb,
                     nullptr, ybufb + (long)DD * NN, DD, NN, DD,
                     DD, 1, NN, 1, NN, 1,
                     BB, 1, 0, 0, DN, 0, (long)512 * NN, 0,
                     gnn_merge_b + (long)i * DD, 1.f);
    // concat first half: copy current descriptor (bf16) into ybuf
    {
      dim3 g((unsigned)((DN + 255) / 256), BB);
      bcopy_b<<<g, 256, 0, stream>>>(ybufb, xb, DN, 512L * NN, DN);
    }
    // mlp1 (f32 out for exact ddof-1 LayerNorm) + LN+ReLU (bf16) + mlp2 (f32)
    launch_gemm<0,0>(stream, mlp1b + (long)i * 512 * 512, ybufb, h1f, nullptr,
                     512, NN, 512, 512, 1, NN, 1, NN, 1,
                     BB, 1, 0, 0, 512L * NN, 0, 512L * NN, 0,
                     gnn_mlp1_b + (long)i * 512, 1.f);
    ln_relu_bf16<<<BB * NN, 256, 0, stream>>>(h1f, gnn_ln_g + (long)i * 512,
                                              gnn_ln_b + (long)i * 512, h1b, 512);
    launch_gemm<0,0>(stream, mlp2b + (long)i * DD * 512, h1b, dxf, nullptr,
                     DD, NN, 512, 512, 1, NN, 1, NN, 1,
                     BB, 1, 0, 0, 512L * NN, 0, DN, 0,
                     gnn_mlp2_b + (long)i * DD, 1.f);
  };

  long cnt = (long)BB * DN;
  for (int i = 0; i < LL; ++i) {
    prop(i, desc0b, desc1b, dx0f);   // both props read pre-update descriptors
    prop(i, desc1b, desc0b, dx1f);
    add_update<<<(unsigned)((cnt + 255) / 256), 256, 0, stream>>>(desc0f, desc0b, dx0f, cnt);
    add_update<<<(unsigned)((cnt + 255) / 256), 256, 0, stream>>>(desc1f, desc1b, dx1f, cnt);
  }

  // 4) final projection (bf16 out) + score matrix
  launch_gemm<0,0>(stream, finalb, desc0b, nullptr, md0b, DD, NN, DD,
                   DD, 1, NN, 1, NN, 1,
                   BB, 1, 0, 0, DN, 0, DN, 0, final_b, 1.f);
  launch_gemm<0,0>(stream, finalb, desc1b, nullptr, md1b, DD, NN, DD,
                   DD, 1, NN, 1, NN, 1,
                   BB, 1, 0, 0, DN, 0, DN, 0, final_b, 1.f);
  // sc[b,n,m] = (1/16) sum_d md0[b,d,n]*md1[b,d,m]; A M-contig (AM=1)
  launch_gemm<1,0>(stream, md0b, md1b, scmat, nullptr, NN, NN, DD,
                   1, NN, NN, 1, NN, 1,
                   BB, 1, DN, 0, DN, 0, (long)NN * NN, 0,
                   nullptr, 1.f / 16.f);

  // 5) Sinkhorn (20 iterations, log domain)
  long tz = (long)BB * M1 * M1;
  build_Z<<<(unsigned)((tz + 255) / 256), 256, 0, stream>>>(scmat, bin_score, Zf);
  zero_f<<<(unsigned)((BB * M1 + 255) / 256), 256, 0, stream>>>(vbuf, (long)BB * M1);
  for (int it = 0; it < 20; ++it) {
    sink_u<<<BB * M1, 256, 0, stream>>>(Zf, vbuf, ubuf);
    sink_v<<<BB * M1, 256, 0, stream>>>(Zf, ubuf, vbuf);
  }
  sink_final<<<(unsigned)((tz + 255) / 256), 256, 0, stream>>>(Zf, ubuf, vbuf, (float*)d_out);
}
